// TFAlbertAttention_55336358642074
// MI455X (gfx1250) — compile-verified, matrix operand order checked by reading, not run on hardware
//
#include <hip/hip_runtime.h>

#define S_LEN  2048
#define DMODEL 1024
#define NHEAD  16
#define DHEAD  64
#define BATCH  4
#define NROWS  (BATCH * S_LEN)   /* 8192 token rows */

typedef _Float16 f16;
typedef __attribute__((ext_vector_type(8)))  _Float16 v8h;
typedef __attribute__((ext_vector_type(16))) _Float16 v16h;
typedef __attribute__((ext_vector_type(8)))  float    v8f;

static __device__ __forceinline__ v16h make16(v8h lo, v8h hi) {
    return __builtin_shufflevector(lo, hi, 0,1,2,3,4,5,6,7,8,9,10,11,12,13,14,15);
}

// Load a 16x32 f16 WMMA operand tile in the A-layout (also used for the B
// operand when the matrix is stored [N][K], i.e. pre-transposed).
// Per ISA 7.12.2 (16-bit A 16x32): lanes 0-15 hold K = k0+0..7 and k0+16..23,
// lanes 16-31 hold K = k0+8..15 and k0+24..31 -> two contiguous 16B chunks.
static __device__ __forceinline__ v16h load_tile(const f16* __restrict__ base,
                                                 int ld, int row0, int k0, int lane) {
    const int hl = lane >> 4;
    const int m  = lane & 15;
    const f16* p = base + (size_t)(row0 + m) * ld + k0 + hl * 8;
    v8h lo = *(const v8h*)p;
    v8h hi = *(const v8h*)(p + 16);
    return make16(lo, hi);
}

static __device__ __forceinline__ v8f wmma_f16(v16h a, v16h b, v8f c) {
    return __builtin_amdgcn_wmma_f32_16x16x32_f16(false, a, false, b, (short)0, c,
                                                  false, false);
}

// ---------------------------------------------------------------------------
// Kernel 0a: f32 -> f16 elementwise convert (activations x)
// ---------------------------------------------------------------------------
__global__ void cvt_f32_to_f16(const float* __restrict__ in, f16* __restrict__ out, int n) {
    int i = blockIdx.x * 256 + threadIdx.x;
    if (i < n) out[i] = (f16)in[i];
}

// ---------------------------------------------------------------------------
// Kernel 0b: W[k][n] (f32, 1024x1024) -> Wt[n][k] (f16), LDS-tiled transpose
// ---------------------------------------------------------------------------
__global__ void transpose_cvt(const float* __restrict__ W, f16* __restrict__ Wt) {
    __shared__ float tile[32][33];
    const int bx = blockIdx.x * 32, by = blockIdx.y * 32;
    const int tx = threadIdx.x, ty = threadIdx.y;
#pragma unroll
    for (int i = 0; i < 32; i += 8)
        tile[ty + i][tx] = W[(size_t)(by + ty + i) * DMODEL + bx + tx];
    __syncthreads();
#pragma unroll
    for (int i = 0; i < 32; i += 8)
        Wt[(size_t)(bx + ty + i) * DMODEL + by + tx] = (f16)tile[tx][ty + i];
}

// ---------------------------------------------------------------------------
// Kernel 1: C = A(8192x1024) * Bt^T + bias   (A f16 row-major, Bt = [N][K] f16)
// One wave -> 16x64 strip.  Ping-pong K pipeline (no register rotation):
//   buf0 computes while buf1 loads, then swap roles.  Peeled tail.
// vtrans=1 stores V transposed per head: Vt[b][h][d][s].
// ---------------------------------------------------------------------------
__global__ void gemm_qkv(const f16* __restrict__ A, const f16* __restrict__ Bt,
                         const float* __restrict__ bias, f16* __restrict__ out,
                         int vtrans) {
    const int lane = threadIdx.x & 31;
    const int wid  = threadIdx.x >> 5;
    const int g    = blockIdx.x * 4 + wid;          // 8192 waves total
    const int m0   = (g >> 4) * 16;                 // 512 M-tiles
    const int n0   = (g & 15) * 64;                 // 16 N-strips of 64
    const int hl   = lane >> 4;

    v8f acc[4];
#pragma unroll
    for (int j = 0; j < 4; ++j)
#pragma unroll
        for (int r = 0; r < 8; ++r) acc[j][r] = 0.f;

    v16h a0, a1, b0[4], b1[4];
    a0 = load_tile(A, DMODEL, m0, 0, lane);
#pragma unroll
    for (int j = 0; j < 4; ++j) b0[j] = load_tile(Bt, DMODEL, n0 + j * 16, 0, lane);

    for (int k = 0; k + 64 < DMODEL; k += 64) {
        __builtin_prefetch(A + (size_t)(m0 + (lane & 15)) * DMODEL + k + 64, 0, 1);
        a1 = load_tile(A, DMODEL, m0, k + 32, lane);
#pragma unroll
        for (int j = 0; j < 4; ++j) b1[j] = load_tile(Bt, DMODEL, n0 + j * 16, k + 32, lane);
#pragma unroll
        for (int j = 0; j < 4; ++j) acc[j] = wmma_f16(a0, b0[j], acc[j]);

        a0 = load_tile(A, DMODEL, m0, k + 64, lane);
#pragma unroll
        for (int j = 0; j < 4; ++j) b0[j] = load_tile(Bt, DMODEL, n0 + j * 16, k + 64, lane);
#pragma unroll
        for (int j = 0; j < 4; ++j) acc[j] = wmma_f16(a1, b1[j], acc[j]);
    }
    // tail: k = DMODEL-64 held in buf0, load DMODEL-32 into buf1
    a1 = load_tile(A, DMODEL, m0, DMODEL - 32, lane);
#pragma unroll
    for (int j = 0; j < 4; ++j) b1[j] = load_tile(Bt, DMODEL, n0 + j * 16, DMODEL - 32, lane);
#pragma unroll
    for (int j = 0; j < 4; ++j) acc[j] = wmma_f16(a0, b0[j], acc[j]);
#pragma unroll
    for (int j = 0; j < 4; ++j) acc[j] = wmma_f16(a1, b1[j], acc[j]);

    // --- bias + store ---
#pragma unroll
    for (int j = 0; j < 4; ++j) {
        const int col = n0 + j * 16 + (lane & 15);
        const float bv = bias[col];
#pragma unroll
        for (int r = 0; r < 8; ++r) {
            const int row = m0 + r + 8 * hl;
            const f16 v = (f16)(acc[j][r] + bv);
            if (!vtrans) {
                out[(size_t)row * DMODEL + col] = v;
            } else {
                const int h = col >> 6, d = col & 63;
                const int bb = row >> 11, s = row & (S_LEN - 1);
                out[(size_t)(((bb * NHEAD + h) * DHEAD + d)) * S_LEN + s] = v;
            }
        }
    }
}

// ---------------------------------------------------------------------------
// Kernel 2: flash-style attention.  One wave = one 16-query tile of one (b,h).
// Phase-disjoint tile lifetimes: K-tiles live only through the score WMMAs;
// V-tiles are issued right after (K dead) and their latency hides under the
// softmax VALU block.  P staged via LDS (C-layout -> A-layout).
// ---------------------------------------------------------------------------
__global__ void attn(const f16* __restrict__ Q, const f16* __restrict__ K,
                     const f16* __restrict__ Vt, const float* __restrict__ mask,
                     f16* __restrict__ ctx) {
    __shared__ f16 plds[4][16 * 64];                // per-wave P staging tile
    const int lane = threadIdx.x & 31;
    const int wid  = threadIdx.x >> 5;
    const int g    = blockIdx.x * 4 + wid;          // 8192 waves total
    const int qt   = g & 127;                       // 128 q-tiles per (b,h)
    const int bh   = g >> 7;
    const int b    = bh >> 4;
    const int h    = bh & 15;
    const int q0   = qt * 16;
    const int hl   = lane >> 4;

    float mrow[8], lrow[8];
    v8f cacc[4];
#pragma unroll
    for (int r = 0; r < 8; ++r) { mrow[r] = -3.0e38f; lrow[r] = 0.f; }
#pragma unroll
    for (int j = 0; j < 4; ++j)
#pragma unroll
        for (int r = 0; r < 8; ++r) cacc[j][r] = 0.f;

    // Q tile is loop-invariant: load both K-steps once.
    v16h qa[2];
#pragma unroll
    for (int kk = 0; kk < 2; ++kk)
        qa[kk] = load_tile(Q, DMODEL, b * S_LEN + q0, h * DHEAD + kk * 32, lane);

    f16* pl = &plds[wid][0];

    for (int kb = 0; kb < S_LEN / 64; ++kb) {
        // ---- scores: load K-tiles (one clause), 8 WMMAs ----
        v8f sc[4];
#pragma unroll
        for (int j = 0; j < 4; ++j)
#pragma unroll
            for (int r = 0; r < 8; ++r) sc[j][r] = 0.f;
        {
            v16h kt[2][4];
#pragma unroll
            for (int kk = 0; kk < 2; ++kk)
#pragma unroll
                for (int j = 0; j < 4; ++j)
                    kt[kk][j] = load_tile(K, DMODEL, b * S_LEN + kb * 64 + j * 16,
                                          h * DHEAD + kk * 32, lane);
#pragma unroll
            for (int kk = 0; kk < 2; ++kk)
#pragma unroll
                for (int j = 0; j < 4; ++j)
                    sc[j] = wmma_f16(qa[kk], kt[kk][j], sc[j]);
        }   // K-tiles dead here

        // ---- issue V-tile loads now; latency hides under softmax ----
        v16h vt[2][4];
#pragma unroll
        for (int kk = 0; kk < 2; ++kk)
#pragma unroll
            for (int j = 0; j < 4; ++j)
                vt[kk][j] = load_tile(Vt, S_LEN, (b * NHEAD + h) * DHEAD + j * 16,
                                      kb * 64 + kk * 32, lane);

        // ---- scale + additive mask (mask is (B,1,1,S) broadcast) ----
#pragma unroll
        for (int j = 0; j < 4; ++j) {
            const int kpos = kb * 64 + j * 16 + (lane & 15);
            const float mv = mask[b * S_LEN + kpos];
#pragma unroll
            for (int r = 0; r < 8; ++r) sc[j][r] = sc[j][r] * 0.125f + mv;
        }

        // ---- online softmax ----
        float rs[8], resc[8];
#pragma unroll
        for (int r = 0; r < 8; ++r) {
            float m = sc[0][r];
#pragma unroll
            for (int j = 1; j < 4; ++j) m = fmaxf(m, sc[j][r]);
#pragma unroll
            for (int x = 1; x < 16; x <<= 1) m = fmaxf(m, __shfl_xor(m, x));
            const float mnew = fmaxf(mrow[r], m);
            resc[r] = __expf(mrow[r] - mnew);
            mrow[r] = mnew;
            rs[r] = 0.f;
        }
#pragma unroll
        for (int j = 0; j < 4; ++j)
#pragma unroll
            for (int r = 0; r < 8; ++r) {
                const float p = __expf(sc[j][r] - mrow[r]);
                sc[j][r] = p;
                rs[r] += p;
            }
#pragma unroll
        for (int r = 0; r < 8; ++r) {
#pragma unroll
            for (int x = 1; x < 16; x <<= 1) rs[r] += __shfl_xor(rs[r], x);
            lrow[r] = lrow[r] * resc[r] + rs[r];
        }
#pragma unroll
        for (int j = 0; j < 4; ++j)
#pragma unroll
            for (int r = 0; r < 8; ++r) cacc[j][r] *= resc[r];

        // ---- P: C-layout regs -> row-major LDS (wave-local, DS in-order) ----
#pragma unroll
        for (int j = 0; j < 4; ++j)
#pragma unroll
            for (int r = 0; r < 8; ++r)
                pl[(r + 8 * hl) * 64 + j * 16 + (lane & 15)] = (f16)sc[j][r];

        // ---- ctx += P (16x64) * V (64x16 per d-tile) ----
#pragma unroll
        for (int kk = 0; kk < 2; ++kk) {
            const f16* pp = pl + (lane & 15) * 64 + kk * 32 + hl * 8;
            v8h lo = *(const v8h*)pp;
            v8h hi = *(const v8h*)(pp + 16);
            v16h pa = make16(lo, hi);
#pragma unroll
            for (int j = 0; j < 4; ++j)
                cacc[j] = wmma_f16(pa, vt[kk][j], cacc[j]);
        }
    }

    // ---- normalize by row sum, store ctx row-major f16 ----
#pragma unroll
    for (int j = 0; j < 4; ++j)
#pragma unroll
        for (int r = 0; r < 8; ++r) {
            const int row = b * S_LEN + q0 + r + 8 * hl;
            const int col = h * DHEAD + j * 16 + (lane & 15);
            ctx[(size_t)row * DMODEL + col] = (f16)(cacc[j][r] / lrow[r]);
        }
}

// ---------------------------------------------------------------------------
// Kernel 3: out = LayerNorm(ctx @ Wo + bo + x) * gamma + beta
// One block = 16 full rows, 512 threads = 16 waves; each wave owns a 64-col
// strip (4 B-tiles -> low VGPR pressure, no spills).  Ping-pong K pipeline.
// LayerNorm: one wave per row, full-wave32 shuffle reduction.
// ---------------------------------------------------------------------------
__global__ void oproj_ln(const f16* __restrict__ Ctx, const f16* __restrict__ WoT,
                         const float* __restrict__ bo, const float* __restrict__ x,
                         const float* __restrict__ gam, const float* __restrict__ bet,
                         float* __restrict__ out) {
    extern __shared__ float ybuf[];                 // [16][1040] padded
    const int YLD  = 1040;
    const int lane = threadIdx.x & 31;
    const int wid  = threadIdx.x >> 5;              // 16 waves
    const int hl   = lane >> 4;
    const int row0 = blockIdx.x * 16;
    const int n0   = wid * 64;

    v8f acc[4];
#pragma unroll
    for (int j = 0; j < 4; ++j)
#pragma unroll
        for (int r = 0; r < 8; ++r) acc[j][r] = 0.f;

    v16h a0, a1, b0[4], b1[4];
    a0 = load_tile(Ctx, DMODEL, row0, 0, lane);
#pragma unroll
    for (int j = 0; j < 4; ++j) b0[j] = load_tile(WoT, DMODEL, n0 + j * 16, 0, lane);

    for (int k = 0; k + 64 < DMODEL; k += 64) {
        __builtin_prefetch(Ctx + (size_t)(row0 + (lane & 15)) * DMODEL + k + 64, 0, 1);
        a1 = load_tile(Ctx, DMODEL, row0, k + 32, lane);
#pragma unroll
        for (int j = 0; j < 4; ++j) b1[j] = load_tile(WoT, DMODEL, n0 + j * 16, k + 32, lane);
#pragma unroll
        for (int j = 0; j < 4; ++j) acc[j] = wmma_f16(a0, b0[j], acc[j]);

        a0 = load_tile(Ctx, DMODEL, row0, k + 64, lane);
#pragma unroll
        for (int j = 0; j < 4; ++j) b0[j] = load_tile(WoT, DMODEL, n0 + j * 16, k + 64, lane);
#pragma unroll
        for (int j = 0; j < 4; ++j) acc[j] = wmma_f16(a1, b1[j], acc[j]);
    }
    a1 = load_tile(Ctx, DMODEL, row0, DMODEL - 32, lane);
#pragma unroll
    for (int j = 0; j < 4; ++j) b1[j] = load_tile(WoT, DMODEL, n0 + j * 16, DMODEL - 32, lane);
#pragma unroll
    for (int j = 0; j < 4; ++j) acc[j] = wmma_f16(a0, b0[j], acc[j]);
#pragma unroll
    for (int j = 0; j < 4; ++j) acc[j] = wmma_f16(a1, b1[j], acc[j]);

    // y = acc + bias + residual  ->  LDS
#pragma unroll
    for (int j = 0; j < 4; ++j) {
        const int col = n0 + j * 16 + (lane & 15);
        const float bv = bo[col];
#pragma unroll
        for (int r = 0; r < 8; ++r) {
            const int row = r + 8 * hl;
            ybuf[row * YLD + col] =
                acc[j][r] + bv + x[(size_t)(row0 + row) * DMODEL + col];
        }
    }
    __syncthreads();

    // LayerNorm: one wave per row, full-wave32 xor-shuffle reduction
    const int row = wid;
    float s = 0.f, s2 = 0.f;
#pragma unroll 4
    for (int j = 0; j < 32; ++j) {
        const float v = ybuf[row * YLD + lane + 32 * j];
        s += v; s2 += v * v;
    }
#pragma unroll
    for (int m = 1; m < 32; m <<= 1) { s += __shfl_xor(s, m); s2 += __shfl_xor(s2, m); }
    const float mu  = s * (1.f / DMODEL);
    const float var = s2 * (1.f / DMODEL) - mu * mu;
    const float inv = rsqrtf(var + 1e-12f);
#pragma unroll 4
    for (int j = 0; j < 32; ++j) {
        const int col = lane + 32 * j;
        const float v = ybuf[row * YLD + col];
        out[(size_t)(row0 + row) * DMODEL + col] = (v - mu) * inv * gam[col] + bet[col];
    }
}

// ---------------------------------------------------------------------------
extern "C" void kernel_launch(void* const* d_in, const int* in_sizes, int n_in,
                              void* d_out, int out_size, void* d_ws, size_t ws_size,
                              hipStream_t stream) {
    const float* x    = (const float*)d_in[0];
    const float* mask = (const float*)d_in[1];
    const float* Wq   = (const float*)d_in[2];
    const float* bq   = (const float*)d_in[3];
    const float* Wk   = (const float*)d_in[4];
    const float* bk   = (const float*)d_in[5];
    const float* Wv   = (const float*)d_in[6];
    const float* bv   = (const float*)d_in[7];
    const float* Wo   = (const float*)d_in[8];
    const float* bo   = (const float*)d_in[9];
    const float* gam  = (const float*)d_in[10];
    const float* bet  = (const float*)d_in[11];
    float* out = (float*)d_out;

    char* ws = (char*)d_ws;
    size_t off = 0;
    f16* xh  = (f16*)(ws + off); off += (size_t)NROWS * DMODEL * 2;
    f16* WqT = (f16*)(ws + off); off += (size_t)DMODEL * DMODEL * 2;
    f16* WkT = (f16*)(ws + off); off += (size_t)DMODEL * DMODEL * 2;
    f16* WvT = (f16*)(ws + off); off += (size_t)DMODEL * DMODEL * 2;
    f16* WoT = (f16*)(ws + off); off += (size_t)DMODEL * DMODEL * 2;
    f16* Qh  = (f16*)(ws + off); off += (size_t)NROWS * DMODEL * 2;
    f16* Kh  = (f16*)(ws + off); off += (size_t)NROWS * DMODEL * 2;
    f16* Vt  = (f16*)(ws + off); off += (size_t)NROWS * DMODEL * 2;
    f16* Ctx = (f16*)(ws + off); off += (size_t)NROWS * DMODEL * 2;

    // 0) pack activations + transpose/convert weights
    {
        const int n = NROWS * DMODEL;
        cvt_f32_to_f16<<<(n + 255) / 256, 256, 0, stream>>>(x, xh, n);
        dim3 tb(32, 8), tg(DMODEL / 32, DMODEL / 32);
        transpose_cvt<<<tg, tb, 0, stream>>>(Wq, WqT);
        transpose_cvt<<<tg, tb, 0, stream>>>(Wk, WkT);
        transpose_cvt<<<tg, tb, 0, stream>>>(Wv, WvT);
        transpose_cvt<<<tg, tb, 0, stream>>>(Wo, WoT);
    }

    // 1) Q/K/V projections (V stored transposed per head)
    {
        const int waves  = (NROWS / 16) * (DMODEL / 64);  // 8192
        const int blocks = waves / 4;
        gemm_qkv<<<blocks, 128, 0, stream>>>(xh, WqT, bq, Qh, 0);
        gemm_qkv<<<blocks, 128, 0, stream>>>(xh, WkT, bk, Kh, 0);
        gemm_qkv<<<blocks, 128, 0, stream>>>(xh, WvT, bv, Vt, 1);
    }

    // 2) attention
    {
        const int waves = BATCH * NHEAD * (S_LEN / 16);   // 8192
        attn<<<waves / 4, 128, 0, stream>>>(Qh, Kh, Vt, mask, Ctx);
    }

    // 3) out-projection + residual + LayerNorm
    {
        const size_t shmem = 16 * 1040 * sizeof(float);   // 66560 B
        oproj_ln<<<NROWS / 16, 512, shmem, stream>>>(Ctx, WoT, bo, x, gam, bet, out);
    }
}
